// LSTMCell2_43834436223774
// MI455X (gfx1250) — matrix operand order repaired and checked by reading, not compiled, not run on hardware
//
#include <hip/hip_runtime.h>

// ---------------------------------------------------------------------------
// Problem constants (from reference): B=4096 rows, H=2048, 4H=8192
// ---------------------------------------------------------------------------
#define MDIM 4096
#define KDIM 2048
#define NDIM 8192
#define EPSV 1e-5f

typedef __attribute__((ext_vector_type(16))) __bf16 v16bf;
typedef __attribute__((ext_vector_type(2)))  __bf16 v2bf;
typedef __attribute__((ext_vector_type(8)))  float  v8f;

struct U8 { uint4 a, b; };  // 32 bytes == v16bf

// Two fp32 -> packed bf16 pair using native converts (v_cvt_pk_bf16_f32 path)
__device__ __forceinline__ unsigned int cvt2(float lo, float hi) {
    v2bf t;
    t.x = (__bf16)lo;
    t.y = (__bf16)hi;
    return __builtin_bit_cast(unsigned int, t);
}

// ---------------------------------------------------------------------------
// fp32 -> bf16 pre-conversion pass (8 elements / thread)
// ---------------------------------------------------------------------------
__global__ __launch_bounds__(256) void
cvt_f32_bf16(const float* __restrict__ src, unsigned short* __restrict__ dst, int n8) {
    const int i = blockIdx.x * 256 + threadIdx.x;
    if (i < n8) {
        const float4 v0 = ((const float4*)src)[2 * i];
        const float4 v1 = ((const float4*)src)[2 * i + 1];
        uint4 o;
        o.x = cvt2(v0.x, v0.y); o.y = cvt2(v0.z, v0.w);
        o.z = cvt2(v1.x, v1.y); o.w = cvt2(v1.z, v1.w);
        ((uint4*)dst)[i] = o;
    }
}

// ---------------------------------------------------------------------------
// Staging: load 16 contiguous K-elements for one tile row into 2 uint4 regs.
// BF16IN: straight 32B bf16 load.  !BF16IN: 64B fp32 load + native convert.
// ---------------------------------------------------------------------------
template <bool BF16IN>
__device__ __forceinline__ void
load_stage(const void* base, long elemOff, uint4& r0, uint4& r1) {
    if constexpr (BF16IN) {
        const unsigned short* p = (const unsigned short*)base + elemOff;
        r0 = ((const uint4*)p)[0];
        r1 = ((const uint4*)p)[1];
    } else {
        const float* p = (const float*)base + elemOff;
        const float4 v0 = ((const float4*)p)[0];
        const float4 v1 = ((const float4*)p)[1];
        const float4 v2 = ((const float4*)p)[2];
        const float4 v3 = ((const float4*)p)[3];
        r0.x = cvt2(v0.x, v0.y); r0.y = cvt2(v0.z, v0.w);
        r0.z = cvt2(v1.x, v1.y); r0.w = cvt2(v1.z, v1.w);
        r1.x = cvt2(v2.x, v2.y); r1.y = cvt2(v2.z, v2.w);
        r1.z = cvt2(v3.x, v3.y); r1.w = cvt2(v3.z, v3.w);
    }
}

// ---------------------------------------------------------------------------
// Z[M,N] = h[M,K] @ W[N,K]^T   (bias deferred to pass 2)
// 128x128 block tile, 8 wave32 waves, each wave: 64x32 = 4x2 WMMA tiles,
// K-step 32 (one v_wmma_f32_16x16x32_bf16 per tile per step).
// LDS rows padded to 40 bf16 (80B = 5 x 16B): bank-conflict relief while
// keeping ds_load_b128 alignment.
// ---------------------------------------------------------------------------
#define LDSTRIDE 40

template <bool BF16IN>
__global__ __launch_bounds__(256) void
gemm_bf16_wmma(const void* __restrict__ Aptr,   // h  [M,K]  (bf16 or fp32)
               const void* __restrict__ Bptr,   // W  [N,K]  (bf16 or fp32)
               float* __restrict__ Z) {
    __shared__ unsigned short As[2][128 * LDSTRIDE];
    __shared__ unsigned short Bs[2][128 * LDSTRIDE];

    const int tid = threadIdx.x;
    const int m0  = blockIdx.y * 128;
    const int n0  = blockIdx.x * 128;

    // staging assignment: each thread covers one (row, 16-wide K chunk)
    const int srow = tid >> 1;
    const int skc  = (tid & 1) * 16;
    const long aOff = (long)(m0 + srow) * KDIM + skc;
    const long bOff = (long)(n0 + srow) * KDIM + skc;

    uint4 ar0, ar1, br0, br1;
    load_stage<BF16IN>(Aptr, aOff, ar0, ar1);
    load_stage<BF16IN>(Bptr, bOff, br0, br1);
    {
        uint4* pa = (uint4*)&As[0][srow * LDSTRIDE + skc];
        uint4* pb = (uint4*)&Bs[0][srow * LDSTRIDE + skc];
        pa[0] = ar0; pa[1] = ar1;
        pb[0] = br0; pb[1] = br1;
    }
    __syncthreads();

    const int lane = tid & 31;
    const int wv   = tid >> 5;      // 0..7
    const int wm   = wv >> 2;       // 0..1  (M half)
    const int wn   = wv & 3;        // 0..3  (N quarter)
    const int hi   = lane >> 4;
    const int lo16 = lane & 15;

    v8f acc[4][2] = {};

    const int NK = KDIM / 32;       // 64 K-steps
    for (int kt = 0; kt < NK; ++kt) {
        const int cur = kt & 1;
        if (kt + 1 < NK) {
            load_stage<BF16IN>(Aptr, aOff + (long)(kt + 1) * 32, ar0, ar1);
            load_stage<BF16IN>(Bptr, bOff + (long)(kt + 1) * 32, br0, br1);
            if (kt + 2 < NK) {      // pull the tile after next toward the WGP
                if constexpr (BF16IN) {
                    __builtin_prefetch((const unsigned short*)Aptr + aOff + (kt + 2) * 32, 0, 3);
                    __builtin_prefetch((const unsigned short*)Bptr + bOff + (kt + 2) * 32, 0, 3);
                } else {
                    __builtin_prefetch((const float*)Aptr + aOff + (kt + 2) * 32, 0, 3);
                    __builtin_prefetch((const float*)Bptr + bOff + (kt + 2) * 32, 0, 3);
                }
            }
        }

        // --- fragments from LDS per documented VGPR striping ---
        v16bf afr[4], bfr[2];
#pragma unroll
        for (int t = 0; t < 4; ++t) {
            // A 16x32 bf16: lanes<16 hold K={0..7,16..23}, lanes>=16 K={8..15,24..31}
            const int r = wm * 64 + t * 16 + lo16;
            const unsigned short* p = &As[cur][r * LDSTRIDE + hi * 8];
            U8 u; u.a = *(const uint4*)p; u.b = *(const uint4*)(p + 16);
            afr[t] = __builtin_bit_cast(v16bf, u);
        }
#pragma unroll
        for (int s = 0; s < 2; ++s) {
            // B 32x16 bf16: lanes<16 hold K={0..7,16..23} of col n=lane, lanes>=16 K={8..15,24..31}
            const int n = wn * 32 + s * 16 + lo16;
            const unsigned short* p = &Bs[cur][n * LDSTRIDE + hi * 8];
            U8 u; u.a = *(const uint4*)p; u.b = *(const uint4*)(p + 16);
            bfr[s] = __builtin_bit_cast(v16bf, u);
        }

#pragma unroll
        for (int t = 0; t < 4; ++t)
#pragma unroll
            for (int s = 0; s < 2; ++s)
                acc[t][s] = __builtin_amdgcn_wmma_f32_16x16x32_bf16(
                    /*neg_a=*/false, afr[t], /*neg_b=*/false, bfr[s],
                    /*c_mod=*/(short)0, acc[t][s],
                    /*reuse_a=*/false, /*reuse_b=*/false);

        if (kt + 1 < NK) {
            uint4* pa = (uint4*)&As[cur ^ 1][srow * LDSTRIDE + skc];
            uint4* pb = (uint4*)&Bs[cur ^ 1][srow * LDSTRIDE + skc];
            pa[0] = ar0; pa[1] = ar1;
            pb[0] = br0; pb[1] = br1;
            __syncthreads();
        }
    }

    // epilogue: C/D layout -> lanes 0-15: M=j, N=lane; lanes 16-31: M=j+8
#pragma unroll
    for (int t = 0; t < 4; ++t)
#pragma unroll
        for (int s = 0; s < 2; ++s) {
            const long r = m0 + wm * 64 + t * 16 + hi * 8;
            const long n = n0 + wn * 32 + s * 16 + lo16;
#pragma unroll
            for (int j = 0; j < 8; ++j)
                Z[(r + j) * (long)NDIM + n] = acc[t][s][j];
        }
}

// ---------------------------------------------------------------------------
// Per-row GroupNorm(4 groups of 2048) + LSTM gate math.
// One block (256 threads) per sample row; z+bias cached in 32KB LDS.
// ---------------------------------------------------------------------------
__device__ __forceinline__ float sigf(float x) {
    return 1.0f / (1.0f + __expf(-x));
}

__global__ __launch_bounds__(256) void
gn_lstm(const float* __restrict__ Z,
        const float* __restrict__ bias,
        const float* __restrict__ cold,
        const float* __restrict__ gnw,
        const float* __restrict__ gnb,
        float* __restrict__ hnew,
        float* __restrict__ cnew) {
    __shared__ float zbuf[NDIM];
    __shared__ float rsum[256];
    __shared__ float rss[256];
    __shared__ float smean[4];
    __shared__ float srstd[4];

    const int r   = blockIdx.x;
    const int tid = threadIdx.x;
    const float* zr = Z + (size_t)r * NDIM;

    for (int g = 0; g < 4; ++g) {
        const int base = g * 2048 + tid * 8;
        float4 v0 = *(const float4*)(zr + base);
        float4 v1 = *(const float4*)(zr + base + 4);
        const float4 b0 = *(const float4*)(bias + base);
        const float4 b1 = *(const float4*)(bias + base + 4);
        v0.x += b0.x; v0.y += b0.y; v0.z += b0.z; v0.w += b0.w;
        v1.x += b1.x; v1.y += b1.y; v1.z += b1.z; v1.w += b1.w;
        *(float4*)(zbuf + base)     = v0;
        *(float4*)(zbuf + base + 4) = v1;

        float s  = v0.x + v0.y + v0.z + v0.w + v1.x + v1.y + v1.z + v1.w;
        float ss = v0.x*v0.x + v0.y*v0.y + v0.z*v0.z + v0.w*v0.w
                 + v1.x*v1.x + v1.y*v1.y + v1.z*v1.z + v1.w*v1.w;
        rsum[tid] = s;
        rss[tid]  = ss;
        __syncthreads();
        for (int st = 128; st > 0; st >>= 1) {
            if (tid < st) { rsum[tid] += rsum[tid + st]; rss[tid] += rss[tid + st]; }
            __syncthreads();
        }
        if (tid == 0) {
            const float mean = rsum[0] * (1.0f / 2048.0f);
            const float var  = rss[0] * (1.0f / 2048.0f) - mean * mean;
            smean[g] = mean;
            srstd[g] = rsqrtf(var + EPSV);
        }
        __syncthreads();
    }

    const float mi = smean[0], ri = srstd[0];
    const float mf = smean[1], rf = srstd[1];
    const float mo = smean[2], ro = srstd[2];
    const float mg = smean[3], rg = srstd[3];

#pragma unroll
    for (int j = 0; j < 8; ++j) {
        const int c = tid * 8 + j;
        const float zi = (zbuf[c]        - mi) * ri * gnw[c]        + gnb[c];
        const float zf = (zbuf[2048 + c] - mf) * rf * gnw[2048 + c] + gnb[2048 + c];
        const float zo = (zbuf[4096 + c] - mo) * ro * gnw[4096 + c] + gnb[4096 + c];
        const float zg = (zbuf[6144 + c] - mg) * rg * gnw[6144 + c] + gnb[6144 + c];
        const float co = cold[(size_t)r * 2048 + c];
        const float cn = sigf(zf) * co + sigf(zi) * tanhf(zg);
        const float hn = sigf(zo) * tanhf(cn);
        hnew[(size_t)r * 2048 + c] = hn;
        cnew[(size_t)r * 2048 + c] = cn;
    }
}

// ---------------------------------------------------------------------------
// Launch
// ---------------------------------------------------------------------------
extern "C" void kernel_launch(void* const* d_in, const int* in_sizes, int n_in,
                              void* d_out, int out_size, void* d_ws, size_t ws_size,
                              hipStream_t stream) {
    // inputs (reference order): x(unused), h, c, W, b, gn_weight, gn_bias
    const float* h   = (const float*)d_in[1];
    const float* c   = (const float*)d_in[2];
    const float* W   = (const float*)d_in[3];
    const float* b   = (const float*)d_in[4];
    const float* gnw = (const float*)d_in[5];
    const float* gnb = (const float*)d_in[6];

    float* out  = (float*)d_out;                       // [h_new | c_new]
    float* Z    = (float*)d_ws;                        // 4096*8192 fp32 = 128MB
    float* hnew = out;
    float* cnew = out + (size_t)MDIM * KDIM;

    const size_t zbytes  = (size_t)MDIM * NDIM * sizeof(float);      // 128MB
    const size_t hbbytes = (size_t)MDIM * KDIM * 2;                  // 16MB
    const size_t wbbytes = (size_t)NDIM * KDIM * 2;                  // 32MB

    dim3 g1(NDIM / 128, MDIM / 128);                   // (64, 32)

    if (ws_size >= zbytes + hbbytes + wbbytes) {
        // Pre-convert h and W to bf16 once; GEMM K-loop stages pure bf16.
        unsigned short* hb = (unsigned short*)((char*)d_ws + zbytes);
        unsigned short* wb = hb + (size_t)MDIM * KDIM;
        const int hn8 = MDIM * KDIM / 8;               // 1,048,576
        const int wn8 = NDIM * KDIM / 8;               // 2,097,152
        cvt_f32_bf16<<<(hn8 + 255) / 256, 256, 0, stream>>>(h, hb, hn8);
        cvt_f32_bf16<<<(wn8 + 255) / 256, 256, 0, stream>>>(W, wb, wn8);
        gemm_bf16_wmma<true><<<g1, 256, 0, stream>>>(hb, wb, Z);
    } else {
        // Fallback: convert on the fly inside the GEMM staging.
        gemm_bf16_wmma<false><<<g1, 256, 0, stream>>>(h, W, Z);
    }

    gn_lstm<<<MDIM, 256, 0, stream>>>(Z, b, c, gnw, gnb, hnew, cnew);
}
